// MACEDiffusionAdapted_1297080123515
// MI455X (gfx1250) — matrix operand order repaired and checked by reading, not compiled, not run on hardware
//
#include <hip/hip_runtime.h>
#include <hip/hip_bf16.h>
#include <hip/hip_fp16.h>

#define NN 10000
#define EE 160000
#define NGR 50
#define NPG 200

typedef __attribute__((ext_vector_type(16))) _Float16 v16h;
typedef __attribute__((ext_vector_type(8)))  _Float16 v8h;
typedef __attribute__((ext_vector_type(8)))  float    v8f;

// fast silu: v_exp_f32 + v_rcp_f32 + mul (avoids div_fixup sequence)
__device__ __forceinline__ float silu_f(float x) {
  return x * __builtin_amdgcn_rcpf(1.f + __expf(-x));
}

#define WMMA_F16(A, B, C) \
  __builtin_amdgcn_wmma_f32_16x16x32_f16(false, (A), false, (B), (short)0, (C), false, false)

// ---- fragment helpers -------------------------------------------------------
// WMMA 16-bit A layout per lane (half = lane>>4): fragment element j holds
// K = (j<8 ? 8*half + j : 16 + 8*half + (j-8)) within a 32-wide K block.
__device__ __forceinline__ v8f load8(const float* p) {
  const float4 a = *(const float4*)p;
  const float4 b = *(const float4*)(p + 4);
  v8f r; r[0]=a.x; r[1]=a.y; r[2]=a.z; r[3]=a.w; r[4]=b.x; r[5]=b.y; r[6]=b.z; r[7]=b.w;
  return r;
}
__device__ __forceinline__ v16h pack2(v8f lo, v8f hi) {
  v16h r;
#pragma unroll
  for (int j = 0; j < 8; ++j) { r[j] = (_Float16)lo[j]; r[8 + j] = (_Float16)hi[j]; }
  return r;
}
__device__ __forceinline__ v16h pack2f(const float* row, int half) {
  return pack2(load8(row + 8 * half), load8(row + 16 + 8 * half));
}
__device__ __forceinline__ v16h pack2h(const _Float16* row, int half) {
  const v8h lo = *(const v8h*)(row + 8 * half);
  const v8h hi = *(const v8h*)(row + 16 + 8 * half);
  v16h r;
#pragma unroll
  for (int j = 0; j < 8; ++j) { r[j] = lo[j]; r[8 + j] = hi[j]; }
  return r;
}
__device__ __forceinline__ v8f silu8(float len, v8f w, v8f b) {
  v8f r;
#pragma unroll
  for (int j = 0; j < 8; ++j) r[j] = silu_f(len * w[j] + b[j]);
  return r;
}
__device__ __forceinline__ v8f bf8(v8f a, v8f s, float w2, float w3) {
  v8f r;
#pragma unroll
  for (int j = 0; j < 8; ++j) {
    float sv = s[j] * 0.0625f, av = a[j] * 0.0625f;   // /AVG_NEI
    r[j] = av * (1.f + w2 * sv + w3 * sv * sv);
  }
  return r;
}

// ---------------------------------------------------------------- simple kernels
__global__ void k_init_pos(const float* __restrict__ src, float* __restrict__ dst) {
  int i = blockIdx.x * blockDim.x + threadIdx.x;
  if (i < NN * 3) dst[i] = src[i];
}

__global__ void k_zero(float* __restrict__ p, int n) {
  int i = blockIdx.x * blockDim.x + threadIdx.x;
  if (i < n) p[i] = 0.f;
}

__global__ void k_embed(const float* __restrict__ na, const float* __restrict__ te,
                        const float* __restrict__ W, const float* __restrict__ b,
                        float* __restrict__ h) {
  int i = blockIdx.x * blockDim.x + threadIdx.x;
  if (i >= NN * 64) return;
  int n = i >> 6, f = i & 63;
  float acc = b[f];
#pragma unroll
  for (int k = 0; k < 16; ++k) acc += na[n * 16 + k] * W[k * 64 + f];
#pragma unroll
  for (int k = 0; k < 16; ++k) acc += te[n * 16 + k] * W[(16 + k) * 64 + f];
  h[i] = acc;
}

__global__ void k_meansum(const float* __restrict__ pos, const int* __restrict__ batch,
                          float* __restrict__ mean) {
  int i = blockIdx.x * blockDim.x + threadIdx.x;
  if (i >= NN * 3) return;
  int n = i / 3, kd = i % 3;
  atomicAdd(&mean[batch[n] * 3 + kd], pos[i]);
}

__global__ void k_output(const float* __restrict__ pos, const float* __restrict__ pos0,
                         const int* __restrict__ batch, const float* __restrict__ mean,
                         float* __restrict__ out) {
  int i = blockIdx.x * blockDim.x + threadIdx.x;
  if (i >= NN * 3) return;
  int n = i / 3, kd = i % 3;
  out[i] = pos[i] - mean[batch[n] * 3 + kd] * (1.f / (float)NPG) - pos0[i];
}

// ---------------------------------------------------------------- edge kernel
// Per wave: 16 edges. Radial 64->128 (2 chained WMMAs x 8 col tiles), fused
// message formation (live lm 0..3 only), fp32 atomics into agg[N,4,64].
// rad_W2 pre-swizzled into WMMA B-fragment layout in LDS; sender features
// staged transposed so per-row values are one 32B vector LDS load.
__global__ __launch_bounds__(256) void k_edge(
    const float* __restrict__ pos, const float* __restrict__ hsrc, int hstride,
    const int* __restrict__ eidx,
    const float* __restrict__ rW1, const float* __restrict__ rb1,
    const float* __restrict__ rW2, const float* __restrict__ rb2,
    float* __restrict__ agg) {
  __shared__ __align__(32) float    shs[8][64 * 16];     // [f][row] per wave, 32KB
  __shared__ __align__(32) _Float16 swt[8 * 2 * 32 * 16];// B frags [t][kb][lane][16], 16KB
  __shared__ float srw1[64], srb1[64], srb2[128];

  const int tid = threadIdx.x;
  // stage rad_W2 into B-fragment layout (f16)
  for (int i = tid; i < 8 * 2 * 32 * 16; i += 256) {
    int idx = i & 15, ln = (i >> 4) & 31, kb = (i >> 9) & 1, t = i >> 10;
    int k = kb * 32 + (idx >> 3) * 16 + (ln >> 4) * 8 + (idx & 7);
    int n = t * 16 + (ln & 15);
    swt[i] = (_Float16)rW2[k * 256 + n];
  }
  if (tid < 64) { srw1[tid] = rW1[tid]; srb1[tid] = rb1[tid]; }
  if (tid < 128) srb2[tid] = rb2[tid];

  const int wv = tid >> 5, lane = tid & 31;
  const int m = lane & 15, half = lane >> 4;
  const int eb = (blockIdx.x * 8 + wv) * 16;
  const int e = eb + m;
  const int snd = eidx[e];
  const int rcv = eidx[EE + e];
  float vx = pos[rcv * 3 + 0] - pos[snd * 3 + 0];
  float vy = pos[rcv * 3 + 1] - pos[snd * 3 + 1];
  float vz = pos[rcv * 3 + 2] - pos[snd * 3 + 2];
  float len = sqrtf(vx * vx + vy * vy + vz * vz) + 1e-8f;
  float inv = __builtin_amdgcn_rcpf(len);
  const float s3 = 1.7320508075688772f;
  float Y1 = s3 * vy * inv, Y2 = s3 * vz * inv, Y3 = s3 * vx * inv;

  // stage hs gather, transposed [f][row]
  for (int i = lane; i < 16 * 64; i += 32) {
    int row = i & 15, f = i >> 4;
    int sr = __shfl(snd, row);
    shs[wv][f * 16 + row] = hsrc[(long)sr * hstride + f];
  }
  __syncthreads();

  // radial hidden in WMMA A layout
  const v16h A0 = pack2(silu8(len, load8(srw1 + 8 * half),      load8(srb1 + 8 * half)),
                        silu8(len, load8(srw1 + 16 + 8 * half), load8(srb1 + 16 + 8 * half)));
  const v16h A1 = pack2(silu8(len, load8(srw1 + 32 + 8 * half), load8(srb1 + 32 + 8 * half)),
                        silu8(len, load8(srw1 + 48 + 8 * half), load8(srb1 + 48 + 8 * half)));

  // hoist per-D-row receiver base pointers / Y values (row = r + 8*half)
  float* p8[8]; float y1v[8], y2v[8], y3v[8];
#pragma unroll
  for (int r = 0; r < 8; ++r) {
    int row = r + half * 8;
    p8[r]  = agg + (long)__shfl(rcv, row) * 256;
    y1v[r] = __shfl(Y1, row);
    y2v[r] = __shfl(Y2, row);
    y3v[r] = __shfl(Y3, row);
  }

  for (int t = 0; t < 8; ++t) {
    const int n = t * 16 + m;
    const v16h B0 = *(const v16h*)&swt[((t * 2 + 0) * 32 + lane) * 16];
    const v16h B1 = *(const v16h*)&swt[((t * 2 + 1) * 32 + lane) * 16];
    v8f c = {};
    c = WMMA_F16(A0, B0, c);
    c = WMMA_F16(A1, B1, c);
    const int f = n & 63;
    const float bias = srb2[n];
    const v8f hsv = load8(&shs[wv][f * 16 + half * 8]);   // rows half*8..+7
    if (t < 4) {                                           // l-block 0 -> lm0 (Y==1)
#pragma unroll
      for (int r = 0; r < 8; ++r)
        atomicAdd(p8[r] + f, (c[r] + bias) * hsv[r]);
    } else {                                               // l-block 1 -> lm1..3
#pragma unroll
      for (int r = 0; r < 8; ++r) {
        float msg = (c[r] + bias) * hsv[r];
        float* dst = p8[r] + f;
        atomicAdd(dst + 64,  msg * y1v[r]);
        atomicAdd(dst + 128, msg * y2v[r]);
        atomicAdd(dst + 192, msg * y3v[r]);
      }
    }
  }
}

// ---------------------------------------------------------------- node product
// Per wave: 4 nodes x 4 lm rows = 16-row tile. nf = Bf @ prod_W (16x64 @ 64x64).
__global__ __launch_bounds__(256) void k_node_prod(
    const float* __restrict__ agg, const float* __restrict__ pW,
    const float* __restrict__ w2p, const float* __restrict__ w3p, int lay,
    float* __restrict__ nf) {
  const int tid = threadIdx.x, wv = tid >> 5, lane = tid & 31;
  const int m = lane & 15, half = lane >> 4;
  const long nb = (long)(blockIdx.x * 8 + wv) * 4;
  if (nb >= NN) return;
  const float w2 = w2p[lay], w3 = w3p[lay];
  const int j = m >> 2, lm = m & 3;
  const float* arow = agg + (nb + j) * 256;       // lm0 row (= s)
  const float* brow = arow + lm * 64;             // this row's lm
  const v16h A0 = pack2(
      bf8(load8(brow + 8 * half),      load8(arow + 8 * half),      w2, w3),
      bf8(load8(brow + 16 + 8 * half), load8(arow + 16 + 8 * half), w2, w3));
  const v16h A1 = pack2(
      bf8(load8(brow + 32 + 8 * half), load8(arow + 32 + 8 * half), w2, w3),
      bf8(load8(brow + 48 + 8 * half), load8(arow + 48 + 8 * half), w2, w3));
  for (int t = 0; t < 4; ++t) {
    const int n = t * 16 + m;
    v16h B0, B1;
#pragma unroll
    for (int v = 0; v < 8; ++v) {
      int kk = (v >> 2) * 16 + half * 8 + 2 * (v & 3);
      B0[2 * v]     = (_Float16)pW[kk * 64 + n];
      B0[2 * v + 1] = (_Float16)pW[(kk + 1) * 64 + n];
      B1[2 * v]     = (_Float16)pW[(kk + 32) * 64 + n];
      B1[2 * v + 1] = (_Float16)pW[(kk + 33) * 64 + n];
    }
    v8f c = {};
    c = WMMA_F16(A0, B0, c);
    c = WMMA_F16(A1, B1, c);
#pragma unroll
    for (int r = 0; r < 8; ++r) {
      int row = r + half * 8;
      nf[(nb + (row >> 2)) * 256 + (row & 3) * 64 + n] = c[r];
    }
  }
}

// ---------------------------------------------------------------- readout + MLP
// Per wave: 16 nodes. scal = silu(sc@W1+b1)@W2+b2, vecout/pos update, then
// h += silu(cat(h,scal)@mW1+mb1)@mW2+mb2. K=128 chains are 4 WMMAs.
__global__ __launch_bounds__(256) void k_node_read(
    const float* __restrict__ nf,
    const float* __restrict__ rW1, const float* __restrict__ rb1,
    const float* __restrict__ rW2, const float* __restrict__ rb2,
    const float* __restrict__ rWv,
    const float* __restrict__ mW1, const float* __restrict__ mb1,
    const float* __restrict__ mW2, const float* __restrict__ mb2,
    float* __restrict__ h, float* __restrict__ pos) {
  __shared__ __align__(32) _Float16 st1[8][16 * 128];
  __shared__ __align__(32) _Float16 ssc[8][16 * 64];
  const int tid = threadIdx.x, wv = tid >> 5, lane = tid & 31;
  const int m = lane & 15, half = lane >> 4;
  const long nb = (long)(blockIdx.x * 8 + wv) * 16;
  if (nb >= NN) return;

  // GEMM1: t1 = silu(sc @ rW1 + rb1), sc = nf row0, 16x64 @ 64x128
  const v16h A0 = pack2f(nf + (nb + m) * 256, half);
  const v16h A1 = pack2f(nf + (nb + m) * 256 + 32, half);
  for (int t = 0; t < 8; ++t) {
    const int n = t * 16 + m;
    v16h B0, B1;
#pragma unroll
    for (int v = 0; v < 8; ++v) {
      int kk = (v >> 2) * 16 + half * 8 + 2 * (v & 3);
      B0[2 * v]     = (_Float16)rW1[kk * 128 + n];
      B0[2 * v + 1] = (_Float16)rW1[(kk + 1) * 128 + n];
      B1[2 * v]     = (_Float16)rW1[(kk + 32) * 128 + n];
      B1[2 * v + 1] = (_Float16)rW1[(kk + 33) * 128 + n];
    }
    v8f c = {};
    c = WMMA_F16(A0, B0, c);
    c = WMMA_F16(A1, B1, c);
#pragma unroll
    for (int r = 0; r < 8; ++r)
      st1[wv][(r + half * 8) * 128 + n] = (_Float16)silu_f(c[r] + rb1[n]);
  }

  // GEMM2: scal = t1 @ rW2 + rb2, K=128
  const v16h C0 = pack2h(&st1[wv][m * 128], half);
  const v16h C1 = pack2h(&st1[wv][m * 128 + 32], half);
  const v16h C2 = pack2h(&st1[wv][m * 128 + 64], half);
  const v16h C3 = pack2h(&st1[wv][m * 128 + 96], half);
  for (int t = 0; t < 4; ++t) {
    const int n = t * 16 + m;
    v16h B0, B1, B2, B3;
#pragma unroll
    for (int v = 0; v < 8; ++v) {
      int kk = (v >> 2) * 16 + half * 8 + 2 * (v & 3);
      B0[2 * v] = (_Float16)rW2[kk * 64 + n];        B0[2 * v + 1] = (_Float16)rW2[(kk + 1) * 64 + n];
      B1[2 * v] = (_Float16)rW2[(kk + 32) * 64 + n]; B1[2 * v + 1] = (_Float16)rW2[(kk + 33) * 64 + n];
      B2[2 * v] = (_Float16)rW2[(kk + 64) * 64 + n]; B2[2 * v + 1] = (_Float16)rW2[(kk + 65) * 64 + n];
      B3[2 * v] = (_Float16)rW2[(kk + 96) * 64 + n]; B3[2 * v + 1] = (_Float16)rW2[(kk + 97) * 64 + n];
    }
    v8f c = {};
    c = WMMA_F16(C0, B0, c);
    c = WMMA_F16(C1, B1, c);
    c = WMMA_F16(C2, B2, c);
    c = WMMA_F16(C3, B3, c);
#pragma unroll
    for (int r = 0; r < 8; ++r)
      ssc[wv][(r + half * 8) * 64 + n] = (_Float16)(c[r] + rb2[n]);
  }

  // vecout (nf rows 1..3 . read_Wv) + pos update
  for (int task = lane; task < 48; task += 32) {
    int node = task / 3, kd = task % 3;
    const float* rowp = nf + (nb + node) * 256 + (1 + kd) * 64;
    float acc = 0.f;
#pragma unroll
    for (int q = 0; q < 8; ++q) {
      const v8f a = load8(rowp + q * 8);
      const v8f w = load8(rWv + q * 8);
#pragma unroll
      for (int jj = 0; jj < 8; ++jj) acc += a[jj] * w[jj];
    }
    pos[(nb + node) * 3 + kd] += acc;
  }

  // MLP GEMM1: hmid = silu(cat(h, scal) @ mW1 + mb1), K=128
  const v16h M0 = pack2f(h + (nb + m) * 64, half);
  const v16h M1 = pack2f(h + (nb + m) * 64 + 32, half);
  const v16h M2 = pack2h(&ssc[wv][m * 64], half);
  const v16h M3 = pack2h(&ssc[wv][m * 64 + 32], half);
  for (int t = 0; t < 4; ++t) {
    const int n = t * 16 + m;
    v16h B0, B1, B2, B3;
#pragma unroll
    for (int v = 0; v < 8; ++v) {
      int kk = (v >> 2) * 16 + half * 8 + 2 * (v & 3);
      B0[2 * v] = (_Float16)mW1[kk * 64 + n];        B0[2 * v + 1] = (_Float16)mW1[(kk + 1) * 64 + n];
      B1[2 * v] = (_Float16)mW1[(kk + 32) * 64 + n]; B1[2 * v + 1] = (_Float16)mW1[(kk + 33) * 64 + n];
      B2[2 * v] = (_Float16)mW1[(kk + 64) * 64 + n]; B2[2 * v + 1] = (_Float16)mW1[(kk + 65) * 64 + n];
      B3[2 * v] = (_Float16)mW1[(kk + 96) * 64 + n]; B3[2 * v + 1] = (_Float16)mW1[(kk + 97) * 64 + n];
    }
    v8f c = {};
    c = WMMA_F16(M0, B0, c);
    c = WMMA_F16(M1, B1, c);
    c = WMMA_F16(M2, B2, c);
    c = WMMA_F16(M3, B3, c);
#pragma unroll
    for (int r = 0; r < 8; ++r)
      st1[wv][(r + half * 8) * 64 + n] = (_Float16)silu_f(c[r] + mb1[n]);
  }

  // MLP GEMM2: h += hmid @ mW2 + mb2
  const v16h H0 = pack2h(&st1[wv][m * 64], half);
  const v16h H1 = pack2h(&st1[wv][m * 64 + 32], half);
  for (int t = 0; t < 4; ++t) {
    const int n = t * 16 + m;
    v16h B0, B1;
#pragma unroll
    for (int v = 0; v < 8; ++v) {
      int kk = (v >> 2) * 16 + half * 8 + 2 * (v & 3);
      B0[2 * v] = (_Float16)mW2[kk * 64 + n];        B0[2 * v + 1] = (_Float16)mW2[(kk + 1) * 64 + n];
      B1[2 * v] = (_Float16)mW2[(kk + 32) * 64 + n]; B1[2 * v + 1] = (_Float16)mW2[(kk + 33) * 64 + n];
    }
    v8f c = {};
    c = WMMA_F16(H0, B0, c);
    c = WMMA_F16(H1, B1, c);
#pragma unroll
    for (int r = 0; r < 8; ++r) {
      int row = r + half * 8;
      h[(nb + row) * 64 + n] += c[r] + mb2[n];
    }
  }
}

// ---------------------------------------------------------------- launch
extern "C" void kernel_launch(void* const* d_in, const int* in_sizes, int n_in,
                              void* d_out, int out_size, void* d_ws, size_t ws_size,
                              hipStream_t stream) {
  const float* positions  = (const float*)d_in[0];
  const float* node_attrs = (const float*)d_in[1];
  const float* time_emb   = (const float*)d_in[2];
  const int*   eidx       = (const int*)d_in[3];
  const int*   batch      = (const int*)d_in[4];
  const float* emb_W  = (const float*)d_in[5];
  const float* emb_b  = (const float*)d_in[6];
  const float* rad_W1 = (const float*)d_in[7];
  const float* rad_b1 = (const float*)d_in[8];
  const float* rad_W2 = (const float*)d_in[9];
  const float* rad_b2 = (const float*)d_in[10];
  const float* prod_W  = (const float*)d_in[11];
  const float* prod_w2 = (const float*)d_in[12];
  const float* prod_w3 = (const float*)d_in[13];
  const float* read_W1 = (const float*)d_in[14];
  const float* read_b1 = (const float*)d_in[15];
  const float* read_W2 = (const float*)d_in[16];
  const float* read_b2 = (const float*)d_in[17];
  const float* read_Wv = (const float*)d_in[18];
  const float* mlp_W1  = (const float*)d_in[19];
  const float* mlp_b1  = (const float*)d_in[20];
  const float* mlp_W2  = (const float*)d_in[21];
  const float* mlp_b2  = (const float*)d_in[22];
  (void)in_sizes; (void)n_in; (void)out_size; (void)ws_size;

  float* ws   = (float*)d_ws;
  float* pos  = ws;                 // N*3
  float* h    = pos + NN * 3;       // N*64
  float* nfb  = h + NN * 64;        // N*4*64 (node_feats rows 0..3)
  float* agg  = nfb + NN * 256;     // N*4*64 (raw segment sums)
  float* mean = agg + NN * 256;     // NGR*3

  k_init_pos<<<(NN * 3 + 255) / 256, 256, 0, stream>>>(positions, pos);
  k_embed<<<(NN * 64 + 255) / 256, 256, 0, stream>>>(node_attrs, time_emb, emb_W, emb_b, h);

  for (int lay = 0; lay < 2; ++lay) {
    k_zero<<<(NN * 256 + 255) / 256, 256, 0, stream>>>(agg, NN * 256);
    const float* hsrc = (lay == 0) ? h : nfb;
    int hstride = (lay == 0) ? 64 : 256;
    k_edge<<<EE / 128, 256, 0, stream>>>(pos, hsrc, hstride, eidx,
        rad_W1 + lay * 64, rad_b1 + lay * 64,
        rad_W2 + lay * 64 * 256, rad_b2 + lay * 256, agg);
    k_node_prod<<<(2500 + 7) / 8, 256, 0, stream>>>(agg, prod_W + lay * 64 * 64,
        prod_w2, prod_w3, lay, nfb);
    k_node_read<<<(625 + 7) / 8, 256, 0, stream>>>(nfb,
        read_W1 + lay * 64 * 128, read_b1 + lay * 128,
        read_W2 + lay * 128 * 64, read_b2 + lay * 64,
        read_Wv + lay * 64,
        mlp_W1 + lay * 128 * 64, mlp_b1 + lay * 64,
        mlp_W2 + lay * 64 * 64, mlp_b2 + lay * 64,
        h, pos);
  }

  k_zero<<<1, 256, 0, stream>>>(mean, NGR * 3);
  k_meansum<<<(NN * 3 + 255) / 256, 256, 0, stream>>>(pos, batch, mean);
  k_output<<<(NN * 3 + 255) / 256, 256, 0, stream>>>(pos, positions, batch, mean,
                                                     (float*)d_out);
}
